// EditModel_2619930050785
// MI455X (gfx1250) — compile-verified, matrix operand order checked by reading, not run on hardware
//
#include <hip/hip_runtime.h>
#include <hip/hip_bf16.h>
#include <math.h>

// ---------------- types ----------------
typedef __bf16 bf16_t;
typedef __attribute__((ext_vector_type(16))) __bf16 v16bf;
typedef __attribute__((ext_vector_type(8)))  __bf16 v8bf;
typedef __attribute__((ext_vector_type(8)))  float  v8f;

#define SXX 34
#define SYY 34
#define BATCH 256
#define INDIM 66
#define KPIN 96          // IN_DIM padded to multiple of 32
#define HH 512           // H
#define OUTD 65          // V + 1
#define OPAD 80          // OUTD padded to multiple of 16
#define BIGNEG (-1000000000.0f)

__device__ __forceinline__ float em_sigmoid(float x) { return 1.0f / (1.0f + __expf(-x)); }

// ---------------- WMMA helpers ----------------
// 16x32 bf16 fragment, row-major source, stride ld (elements).
// Per CDNA5 ISA 7.12.2 (16-bit A-matrix 16x32): lane L holds row M=L%16,
// K = [ (L/16)*8 .. +8 ) in elems 0..7 and [16+(L/16)*8 .. +8) in elems 8..15.
__device__ __forceinline__ v16bf em_load_frag(const bf16_t* __restrict__ base, int ld) {
    int lane = threadIdx.x & 31;
    int row  = lane & 15;
    int ko   = (lane >> 4) << 3;            // 0 or 8
    const bf16_t* p = base + row * ld + ko;
    v8bf lo = *reinterpret_cast<const v8bf*>(p);
    v8bf hi = *reinterpret_cast<const v8bf*>(p + 16);
    v16bf f;
#pragma unroll
    for (int e = 0; e < 8; ++e) { f[e] = lo[e]; f[e + 8] = hi[e]; }
    return f;
}

__device__ __forceinline__ v8f em_wmma(v16bf a, v16bf b, v8f c) {
    // D = A(16x32) * B(32x16) + C, fp32 accumulate
    return __builtin_amdgcn_wmma_f32_16x16x32_bf16(false, a, false, b, (short)0, c,
                                                   false, false);
}

__device__ __forceinline__ v8f em_zero8() {
    v8f z = {0.f, 0.f, 0.f, 0.f, 0.f, 0.f, 0.f, 0.f};
    return z;
}

// ---------------- weight conversion fp32 -> padded bf16 [Npad, Kpad] ----------------
__global__ void em_convert_pad(const float* __restrict__ in, bf16_t* __restrict__ out,
                               int N, int K, int Npad, int Kpad) {
    int idx = blockIdx.x * blockDim.x + threadIdx.x;
    int total = Npad * Kpad;
    if (idx >= total) return;
    int n = idx / Kpad, k = idx % Kpad;
    float v = (n < N && k < K) ? in[n * K + k] : 0.0f;
    out[idx] = (bf16_t)v;
}

// ---------------- prep: one-hot inputs -> padded bf16, masks, gathers ----------------
__global__ void em_prep(const float* __restrict__ src, const float* __restrict__ tgt,
                        bf16_t* __restrict__ xsrc, bf16_t* __restrict__ ytgt,
                        int* __restrict__ xany, int* __restrict__ yany,
                        int* __restrict__ ysym, int* __restrict__ ypar) {
    int idx = blockIdx.x * blockDim.x + threadIdx.x;   // p*BATCH + b
    if (idx >= SXX * BATCH) return;
    const float* srow = src + (size_t)idx * INDIM;
    const float* trow = tgt + (size_t)idx * INDIM;
    float ssum = 0.f, tsum = 0.f;
#pragma unroll 6
    for (int k = 0; k < INDIM; ++k) { ssum += srow[k]; tsum += trow[k]; }
#pragma unroll 6
    for (int k = 0; k < KPIN; ++k) {
        xsrc[(size_t)idx * KPIN + k] = (bf16_t)(k < INDIM ? srow[k] : 0.0f);
        ytgt[(size_t)idx * KPIN + k] = (bf16_t)(k < INDIM ? trow[k] : 0.0f);
    }
    xany[idx] = ssum > 0.f;
    yany[idx] = tsum > 0.f;
    int p = idx / BATCH;
    int sym = -1, par = 0;
    if (p < SYY - 1) {
        const float* nrow = tgt + (size_t)(idx + BATCH) * INDIM;
        for (int v = 0; v < OUTD - 1; ++v)
            if (nrow[v] > 0.5f) sym = v;
        par = nrow[INDIM - 1] > 0.5f;
    }
    ysym[idx] = sym;
    ypar[idx] = par;
}

// ---------------- fused LSTM cell update for one 16-row half-tile ----------------
__device__ __forceinline__ void em_cell_update(v8f ai, v8f af, v8f ag, v8f ao,
                                               int mbase, int n, int Hp,
                                               float bi, float bf_, float bg, float bo,
                                               float* __restrict__ c,
                                               bf16_t* __restrict__ hout, int first) {
    int lane = threadIdx.x & 31;
#pragma unroll
    for (int r = 0; r < 8; ++r) {
        int mm = mbase + r + ((lane >> 4) << 3);
        float gi = ai[r] + bi, gf = af[r] + bf_, gg = ag[r] + bg, go = ao[r] + bo;
        float cp = first ? 0.0f : c[(size_t)mm * Hp + n];
        float cv = em_sigmoid(gf) * cp + em_sigmoid(gi) * tanhf(gg);
        c[(size_t)mm * Hp + n]    = cv;
        hout[(size_t)mm * Hp + n] = (bf16_t)(em_sigmoid(go) * tanhf(cv));
    }
}

// One wave computes a 32(M) x 16(N) tile of all four gates, fused with cell update.
__device__ __forceinline__ void em_lstm_tile(int wl, int Hp,
                                             const bf16_t* __restrict__ x,
                                             const bf16_t* __restrict__ Wih,
                                             const bf16_t* __restrict__ hprev,
                                             const bf16_t* __restrict__ Whh,
                                             const float*  __restrict__ bias,
                                             float* __restrict__ c,
                                             bf16_t* __restrict__ hout, int first) {
    int ntiles = Hp >> 4;
    int m  = (wl / ntiles) << 5;     // 32-row batch tile (BATCH/32 = 8 tiles)
    int nh = (wl % ntiles) << 4;     // hidden-unit tile

    v8f a0i = em_zero8(), a0f = em_zero8(), a0g = em_zero8(), a0o = em_zero8();
    v8f a1i = em_zero8(), a1f = em_zero8(), a1g = em_zero8(), a1o = em_zero8();

#pragma unroll
    for (int k = 0; k < KPIN; k += 32) {
        v16bf x0 = em_load_frag(x + (size_t)m * KPIN + k, KPIN);
        v16bf x1 = em_load_frag(x + (size_t)(m + 16) * KPIN + k, KPIN);
        v16bf b;
        b = em_load_frag(Wih + (size_t)(0 * Hp + nh) * KPIN + k, KPIN);
        a0i = em_wmma(x0, b, a0i); a1i = em_wmma(x1, b, a1i);
        b = em_load_frag(Wih + (size_t)(1 * Hp + nh) * KPIN + k, KPIN);
        a0f = em_wmma(x0, b, a0f); a1f = em_wmma(x1, b, a1f);
        b = em_load_frag(Wih + (size_t)(2 * Hp + nh) * KPIN + k, KPIN);
        a0g = em_wmma(x0, b, a0g); a1g = em_wmma(x1, b, a1g);
        b = em_load_frag(Wih + (size_t)(3 * Hp + nh) * KPIN + k, KPIN);
        a0o = em_wmma(x0, b, a0o); a1o = em_wmma(x1, b, a1o);
    }
    if (!first) {
        for (int k = 0; k < Hp; k += 32) {
            v16bf h0 = em_load_frag(hprev + (size_t)m * Hp + k, Hp);
            v16bf h1 = em_load_frag(hprev + (size_t)(m + 16) * Hp + k, Hp);
            v16bf b;
            b = em_load_frag(Whh + (size_t)(0 * Hp + nh) * Hp + k, Hp);
            a0i = em_wmma(h0, b, a0i); a1i = em_wmma(h1, b, a1i);
            b = em_load_frag(Whh + (size_t)(1 * Hp + nh) * Hp + k, Hp);
            a0f = em_wmma(h0, b, a0f); a1f = em_wmma(h1, b, a1f);
            b = em_load_frag(Whh + (size_t)(2 * Hp + nh) * Hp + k, Hp);
            a0g = em_wmma(h0, b, a0g); a1g = em_wmma(h1, b, a1g);
            b = em_load_frag(Whh + (size_t)(3 * Hp + nh) * Hp + k, Hp);
            a0o = em_wmma(h0, b, a0o); a1o = em_wmma(h1, b, a1o);
        }
    }

    // C/D layout: vgpr r, lanes 0-15: M=r, N=lane; lanes 16-31: M=8+r, N=lane-16
    int lane = threadIdx.x & 31;
    int n    = nh + (lane & 15);
    float bi = bias[0 * Hp + n], bf_ = bias[1 * Hp + n];
    float bg = bias[2 * Hp + n], bo  = bias[3 * Hp + n];
    em_cell_update(a0i, a0f, a0g, a0o, m,      n, Hp, bi, bf_, bg, bo, c, hout, first);
    em_cell_update(a1i, a1f, a1g, a1o, m + 16, n, Hp, bi, bf_, bg, bo, c, hout, first);
}

// One launch per timestep drives all three independent LSTMs concurrently.
// Waves [0,256): fwd prior (Hp=512); [256,512): rev prior; [512,1024): modern (Hp=1024)
__global__ void em_lstm_step3(
    const bf16_t* xf, const bf16_t* Wih_f, const bf16_t* hpf, const bf16_t* Whh_f,
    const float* bf, float* cf, bf16_t* hof,
    const bf16_t* xr, const bf16_t* Wih_r, const bf16_t* hpr, const bf16_t* Whh_r,
    const float* br, float* cr, bf16_t* hor,
    const bf16_t* xm, const bf16_t* Wih_m, const bf16_t* hpm, const bf16_t* Whh_m,
    const float* bm, float* cm, bf16_t* hom,
    int first) {
    int wave = (blockIdx.x * blockDim.x + threadIdx.x) >> 5;
    if (wave < 256) {
        em_lstm_tile(wave, HH, xf, Wih_f, hpf, Whh_f, bf, cf, hof, first);
    } else if (wave < 512) {
        em_lstm_tile(wave - 256, HH, xr, Wih_r, hpr, Whh_r, br, cr, hor, first);
    } else {
        em_lstm_tile(wave - 512, 2 * HH, xm, Wih_m, hpm, Whh_m, bm, cm, hom, first);
    }
}

// ---------------- head GEMM: M=32 per wave, two heads share A fragments ----------------
// out[M, OPAD] = A[M, Ktot] @ W^T (+bias), A split at K0 between A0/A1
__global__ void em_head_gemm(const bf16_t* __restrict__ A0, int ld0,
                             const bf16_t* __restrict__ A1, int ld1,
                             int K0, int Ktot,
                             const bf16_t* __restrict__ Wa,   // [OPAD, 2H]
                             const bf16_t* __restrict__ Wb,   // [OPAD, 2H]
                             const float* __restrict__ ba, const float* __restrict__ bb,
                             int addBias,
                             float* __restrict__ outa, float* __restrict__ outb) {
    int wave = (blockIdx.x * blockDim.x + threadIdx.x) >> 5;
    const int NT = OPAD / 16;                  // 5 N tiles
    int m  = (wave / NT) << 5;                 // 32-row tile (8704/32 = 272 tiles)
    int nb = (wave % NT) << 4;

    v8f aA0 = em_zero8(), aB0 = em_zero8(), aA1 = em_zero8(), aB1 = em_zero8();
    for (int k = 0; k < Ktot; k += 32) {
        v16bf f0, f1;
        if (k < K0) {
            f0 = em_load_frag(A0 + (size_t)m * ld0 + k, ld0);
            f1 = em_load_frag(A0 + (size_t)(m + 16) * ld0 + k, ld0);
        } else {
            f0 = em_load_frag(A1 + (size_t)m * ld1 + (k - K0), ld1);
            f1 = em_load_frag(A1 + (size_t)(m + 16) * ld1 + (k - K0), ld1);
        }
        v16bf wa = em_load_frag(Wa + (size_t)nb * Ktot + k, Ktot);
        v16bf wb = em_load_frag(Wb + (size_t)nb * Ktot + k, Ktot);
        aA0 = em_wmma(f0, wa, aA0); aA1 = em_wmma(f1, wa, aA1);
        aB0 = em_wmma(f0, wb, aB0); aB1 = em_wmma(f1, wb, aB1);
    }

    int lane = threadIdx.x & 31;
    int n    = nb + (lane & 15);
    float bva = 0.f, bvb = 0.f;
    if (addBias && n < OUTD) { bva = ba[n]; bvb = bb[n]; }
#pragma unroll
    for (int r = 0; r < 8; ++r) {
        int mm = m + r + ((lane >> 4) << 3);
        outa[(size_t)mm * OPAD + n]        = aA0[r] + bva;
        outb[(size_t)mm * OPAD + n]        = aB0[r] + bvb;
        outa[(size_t)(mm + 16) * OPAD + n] = aA1[r] + bva;
        outb[(size_t)(mm + 16) * OPAD + n] = aB1[r] + bvb;
    }
}

// ---------------- pairwise logsumexp + gather + masks ----------------
__global__ void em_pairwise(const float* __restrict__ lxs, const float* __restrict__ lxi,
                            const float* __restrict__ lys, const float* __restrict__ lyi,
                            const int* __restrict__ xany, const int* __restrict__ yany,
                            const int* __restrict__ ysym, const int* __restrict__ ypar,
                            float* __restrict__ out) {
    __shared__ float sxs[SXX][OUTD + 1];
    __shared__ float sxi[SXX][OUTD + 1];
    __shared__ float sys_[SYY][OUTD + 1];
    __shared__ float syi[SYY][OUTD + 1];
    __shared__ int s_xany[SXX], s_yany[SYY], s_sym[SYY], s_par[SYY];

    int b   = blockIdx.x;
    int tid = threadIdx.x;
    for (int idx = tid; idx < SXX * OUTD; idx += blockDim.x) {
        int i = idx / OUTD, o = idx % OUTD;
        size_t g = (size_t)(i * BATCH + b) * OPAD + o;
        sxs[i][o]  = lxs[g];
        sxi[i][o]  = lxi[g];
        sys_[i][o] = lys[g];
        syi[i][o]  = lyi[g];
    }
    if (tid < SXX) {
        s_xany[tid] = xany[tid * BATCH + b];
        s_yany[tid] = yany[tid * BATCH + b];
        s_sym[tid]  = ysym[tid * BATCH + b];
        s_par[tid]  = ypar[tid * BATCH + b];
    }
    __syncthreads();

    const int CH = SXX * SYY * BATCH;
    for (int cell = tid; cell < SXX * SYY; cell += blockDim.x) {
        int i = cell / SYY, j = cell % SYY;
        bool valid = s_xany[i] && s_yany[j] && (j < SYY - 1);

        float m1 = -1e30f, m2 = -1e30f;
#pragma unroll 5
        for (int o = 0; o < OUTD; ++o) {
            float v1 = sxs[i][o] + sys_[j][o];
            float v2 = sxi[i][o] + syi[j][o];
            m1 = fmaxf(m1, v1);
            m2 = fmaxf(m2, v2);
        }
        float s1 = 0.f, s2 = 0.f;
#pragma unroll 5
        for (int o = 0; o < OUTD; ++o) {
            s1 += __expf(sxs[i][o] + sys_[j][o] - m1);
            s2 += __expf(sxi[i][o] + syi[j][o] - m2);
        }
        float lse_s = m1 + __logf(s1);
        float lse_i = m2 + __logf(s2);

        int   sym  = s_sym[j];
        float insv = (sym >= 0) ? (sxi[i][sym] + syi[j][sym] - lse_i) : 0.0f;
        float subv = (sym >= 0) ? (sxs[i][sym] + sys_[j][sym] - lse_s) : 0.0f;
        bool  ins_ok = valid && !s_par[j];

        int base = i * (SYY * BATCH) + j * BATCH + b;
        out[0 * CH + base] = ins_ok ? insv : BIGNEG;
        out[1 * CH + base] = ins_ok ? subv : BIGNEG;
        out[2 * CH + base] = valid ? (sxi[i][OUTD - 1] + syi[j][OUTD - 1] - lse_i) : BIGNEG;
        out[3 * CH + base] = valid ? (sxs[i][OUTD - 1] + sys_[j][OUTD - 1] - lse_s) : BIGNEG;
    }
}

// ---------------- host launcher ----------------
extern "C" void kernel_launch(void* const* d_in, const int* in_sizes, int n_in,
                              void* d_out, int out_size, void* d_ws, size_t ws_size,
                              hipStream_t stream) {
    (void)in_sizes; (void)n_in; (void)out_size; (void)ws_size;
    const float* src    = (const float*)d_in[0];
    const float* tgt    = (const float*)d_in[1];
    const float* Wih_f  = (const float*)d_in[2];
    const float* Whh_f  = (const float*)d_in[3];
    const float* b_f    = (const float*)d_in[4];
    const float* Wih_r  = (const float*)d_in[5];
    const float* Whh_r  = (const float*)d_in[6];
    const float* b_r    = (const float*)d_in[7];
    const float* Wih_m  = (const float*)d_in[8];
    const float* Whh_m  = (const float*)d_in[9];
    const float* b_m    = (const float*)d_in[10];
    const float* W_sub  = (const float*)d_in[11];
    const float* b_sub  = (const float*)d_in[12];
    const float* W_ins  = (const float*)d_in[13];
    const float* b_ins  = (const float*)d_in[14];

    char*  ws  = (char*)d_ws;
    size_t off = 0;
    auto alloc = [&](size_t bytes) -> char* {
        char* p = ws + off;
        off = (off + bytes + 255) & ~((size_t)255);
        return p;
    };

    bf16_t* xsrc  = (bf16_t*)alloc((size_t)SXX * BATCH * KPIN * 2);
    bf16_t* ytgt  = (bf16_t*)alloc((size_t)SYY * BATCH * KPIN * 2);
    bf16_t* wih_f = (bf16_t*)alloc((size_t)4 * HH * KPIN * 2);
    bf16_t* whh_f = (bf16_t*)alloc((size_t)4 * HH * HH * 2);
    bf16_t* wih_r = (bf16_t*)alloc((size_t)4 * HH * KPIN * 2);
    bf16_t* whh_r = (bf16_t*)alloc((size_t)4 * HH * HH * 2);
    bf16_t* wih_m = (bf16_t*)alloc((size_t)8 * HH * KPIN * 2);
    bf16_t* whh_m = (bf16_t*)alloc((size_t)8 * HH * (2 * HH) * 2);
    bf16_t* wsub  = (bf16_t*)alloc((size_t)OPAD * (2 * HH) * 2);
    bf16_t* wins  = (bf16_t*)alloc((size_t)OPAD * (2 * HH) * 2);
    bf16_t* hs_f  = (bf16_t*)alloc((size_t)SXX * BATCH * HH * 2);
    bf16_t* hs_b  = (bf16_t*)alloc((size_t)SXX * BATCH * HH * 2);
    bf16_t* hs_m  = (bf16_t*)alloc((size_t)SYY * BATCH * (2 * HH) * 2);
    float*  c_f   = (float*)alloc((size_t)BATCH * HH * 4);
    float*  c_r   = (float*)alloc((size_t)BATCH * HH * 4);
    float*  c_m   = (float*)alloc((size_t)BATCH * (2 * HH) * 4);
    float*  lxs   = (float*)alloc((size_t)SXX * BATCH * OPAD * 4);
    float*  lxi   = (float*)alloc((size_t)SXX * BATCH * OPAD * 4);
    float*  lys   = (float*)alloc((size_t)SYY * BATCH * OPAD * 4);
    float*  lyi   = (float*)alloc((size_t)SYY * BATCH * OPAD * 4);
    int*    xany  = (int*)alloc((size_t)SXX * BATCH * 4);
    int*    yany  = (int*)alloc((size_t)SYY * BATCH * 4);
    int*    ysym  = (int*)alloc((size_t)SYY * BATCH * 4);
    int*    ypar  = (int*)alloc((size_t)SYY * BATCH * 4);

    // 1) prep inputs + masks
    em_prep<<<(SXX * BATCH + 255) / 256, 256, 0, stream>>>(src, tgt, xsrc, ytgt,
                                                           xany, yany, ysym, ypar);

    // 2) weight conversion fp32 -> bf16 padded
    auto conv = [&](const float* in, bf16_t* outp, int N, int K, int Npad, int Kpad) {
        int total = Npad * Kpad;
        em_convert_pad<<<(total + 255) / 256, 256, 0, stream>>>(in, outp, N, K, Npad, Kpad);
    };
    conv(Wih_f, wih_f, 4 * HH, INDIM, 4 * HH, KPIN);
    conv(Whh_f, whh_f, 4 * HH, HH, 4 * HH, HH);
    conv(Wih_r, wih_r, 4 * HH, INDIM, 4 * HH, KPIN);
    conv(Whh_r, whh_r, 4 * HH, HH, 4 * HH, HH);
    conv(Wih_m, wih_m, 8 * HH, INDIM, 8 * HH, KPIN);
    conv(Whh_m, whh_m, 8 * HH, 2 * HH, 8 * HH, 2 * HH);
    conv(W_sub, wsub, OUTD, 2 * HH, OPAD, 2 * HH);
    conv(W_ins, wins, OUTD, 2 * HH, OPAD, 2 * HH);

    // 3) sequential LSTM steps: one fused launch per timestep drives all three LSTMs.
    // Waves: 256 (fwd) + 256 (rev) + 512 (modern) = 1024 -> 128 blocks of 8 waves.
    for (int t = 0; t < SXX; ++t) {
        int first = (t == 0);
        int si = SXX - 1 - t;   // reverse LSTM walks the sequence backwards
        em_lstm_step3<<<128, 256, 0, stream>>>(
            // forward prior LSTM
            xsrc + (size_t)t * BATCH * KPIN, wih_f,
            first ? hs_f : hs_f + (size_t)(t - 1) * BATCH * HH, whh_f,
            b_f, c_f, hs_f + (size_t)t * BATCH * HH,
            // reverse prior LSTM
            xsrc + (size_t)si * BATCH * KPIN, wih_r,
            first ? hs_b : hs_b + (size_t)(si + 1) * BATCH * HH, whh_r,
            b_r, c_r, hs_b + (size_t)si * BATCH * HH,
            // modern LSTM over targets (hidden 2H)
            ytgt + (size_t)t * BATCH * KPIN, wih_m,
            first ? hs_m : hs_m + (size_t)(t - 1) * BATCH * (2 * HH), whh_m,
            b_m, c_m, hs_m + (size_t)t * BATCH * (2 * HH),
            first);
    }

    // 4) head GEMMs: waves = (8704/32) * (80/16) = 1360 -> 170 blocks
    int hblocks = ((SXX * BATCH / 32) * (OPAD / 16)) / 8;
    // x-side: A = concat(hs_f, hs_b) over features, no bias
    em_head_gemm<<<hblocks, 256, 0, stream>>>(hs_f, HH, hs_b, HH, HH, 2 * HH,
                                              wsub, wins, b_sub, b_ins, 0, lxs, lxi);
    // y-side: A = hs_m, with bias
    em_head_gemm<<<hblocks, 256, 0, stream>>>(hs_m, 2 * HH, hs_m, 2 * HH, 2 * HH, 2 * HH,
                                              wsub, wins, b_sub, b_ins, 1, lys, lyi);

    // 5) pairwise log-softmax reduction + masking
    em_pairwise<<<BATCH, 256, 0, stream>>>(lxs, lxi, lys, lyi, xany, yany, ysym, ypar,
                                           (float*)d_out);
}
